// MultiHeadAttention_76373108457727
// MI455X (gfx1250) — matrix operand order repaired and checked by reading, not compiled, not run on hardware
//
#include <hip/hip_runtime.h>
#include <hip/hip_bf16.h>

// MHA forward for B=4,S=2048,D=512,H=8,HD=64 using CDNA5 WMMA f16 (f32 acc).
// Pipeline: fp32->f16 convert, fused QKV GEMM (V stored transposed),
// transposed flash attention (S^T = K Q^T so softmax is a lane-pair reduce
// and P^T lands in WMMA B-fragment layout), output projection GEMM.
// Workspace use: ~44 MB.

typedef __attribute__((ext_vector_type(16))) _Float16     v16h;
typedef __attribute__((ext_vector_type(8)))  _Float16     v8h;
typedef __attribute__((ext_vector_type(4)))  _Float16     v4h;
typedef __attribute__((ext_vector_type(8)))  float        v8f;
typedef __attribute__((ext_vector_type(8)))  unsigned int v8u;

#define B_  4
#define S_  2048
#define D_  512
#define H_  8
#define HD_ 64

// D = A x B + C, 16x16x32 f16 -> f32
__device__ __forceinline__ v8f wmma_f16(v16h a, v16h b, v8f c) {
  return __builtin_amdgcn_wmma_f32_16x16x32_f16(false, a, false, b, (short)0, c, false, false);
}

// A-fragment (16x32, MxK) from row-major f16 matrix.
// lane 0-15 : row = row0+lane,    K in {k0+0..15}
// lane 16-31: row = row0+lane-16, K in {k0+16..31}
__device__ __forceinline__ v16h load_fragA(const _Float16* __restrict__ base, int ld,
                                           int row0, int k0, int lane) {
  int row = row0 + (lane & 15);
  int kb  = k0 + ((lane >> 4) << 4);
  const _Float16* p = base + (size_t)row * ld + kb;
  v8h lo = *(const v8h*)p;
  v8h hi = *(const v8h*)(p + 8);
  return __builtin_shufflevector(lo, hi, 0,1,2,3,4,5,6,7,8,9,10,11,12,13,14,15);
}

// B-fragment (32x16, KxN) where column n of B = row n of the row-major source.
// lane 0-15 : col = col0+lane, K = k0+0..15 ; lane 16-31: K = k0+16..31 (contiguous)
__device__ __forceinline__ v16h load_fragB(const _Float16* __restrict__ base, int ld,
                                           int col0, int k0, int lane) {
  int row = col0 + (lane & 15);
  int kb  = k0 + ((lane >> 4) << 4);
  return *(const v16h*)(base + (size_t)row * ld + kb);
}

// ---------------------------------------------------------------- convert
__global__ void cvt4_f32_f16(const float* __restrict__ src, _Float16* __restrict__ dst, int n) {
  int i = (blockIdx.x * blockDim.x + threadIdx.x) * 4;
  if (i + 3 < n) {
    float4 v = *(const float4*)(src + i);
    v4h o;
    o[0] = (_Float16)v.x; o[1] = (_Float16)v.y;
    o[2] = (_Float16)v.z; o[3] = (_Float16)v.w;
    *(v4h*)(dst + i) = o;
  }
}

// ---------------------------------------------------------------- fused QKV
// 12288 wave-jobs: proj(3) x m-tile(512) x n-tile(8); each wave -> 16x64 tile.
__global__ void qkv_kernel(const _Float16* __restrict__ xh,
                           const _Float16* __restrict__ Wqh,
                           const _Float16* __restrict__ Wkh,
                           const _Float16* __restrict__ Wvh,
                           const float* __restrict__ bq,
                           const float* __restrict__ bk,
                           const float* __restrict__ bv,
                           _Float16* __restrict__ Q,   // [B,H,S,HD]
                           _Float16* __restrict__ K,   // [B,H,S,HD]
                           _Float16* __restrict__ Vt)  // [B,H,HD,S]
{
  const int lane = threadIdx.x & 31;
  const int job  = blockIdx.x * 8 + (threadIdx.x >> 5);
  const int proj = job >> 12;        // 0=Q 1=K 2=V
  const int rem  = job & 4095;
  const int m0   = (rem >> 3) << 4;  // 512 m-tiles
  const int n0   = (rem & 7) << 6;   // 8 n-tiles of 64

  const _Float16* W    = proj == 0 ? Wqh : (proj == 1 ? Wkh : Wvh);
  const float*    bias = proj == 0 ? bq  : (proj == 1 ? bk  : bv);

  v8f acc[4] = {};
  for (int k0 = 0; k0 < D_; k0 += 32) {
    v16h a = load_fragA(xh, D_, m0, k0, lane);
#pragma unroll
    for (int j = 0; j < 4; ++j) {
      v16h bf = load_fragB(W, D_, n0 + j * 16, k0, lane);
      acc[j] = wmma_f16(a, bf, acc[j]);
    }
  }

  const float qscale = 0.125f * 1.44269504088896f;  // 1/sqrt(HD) * log2(e)
  const int mrow = m0 + ((lane >> 4) << 3);         // rows mrow..mrow+7, same batch
  const int bidx = mrow / S_;
  const int srow = mrow % S_;

#pragma unroll
  for (int j = 0; j < 4; ++j) {
    const int f  = n0 + j * 16 + (lane & 15);
    const int h  = f >> 6;
    const int hd = f & 63;
    const float bb = bias[f];
    if (proj == 2) {
      // V transposed: consecutive r -> consecutive s -> contiguous b128 store
      v8h pk;
#pragma unroll
      for (int r = 0; r < 8; ++r) pk[r] = (_Float16)(acc[j][r] + bb);
      _Float16* dst = Vt + (((size_t)bidx * H_ + h) * HD_ + hd) * S_ + srow;
      *(v8h*)dst = pk;
    } else {
      const float sc = (proj == 0) ? qscale : 1.0f;
      _Float16* dst = (proj == 0 ? Q : K) + (((size_t)bidx * H_ + h) * S_ + srow) * HD_ + hd;
#pragma unroll
      for (int r = 0; r < 8; ++r) dst[(size_t)r * HD_] = (_Float16)((acc[j][r] + bb) * sc);
    }
  }
}

// ---------------------------------------------------------------- attention
// 4096 wave-jobs: (b,h,q-tile of 16). Streams causal k-chunks of 32.
__global__ void attn_kernel(const _Float16* __restrict__ Q,
                            const _Float16* __restrict__ K,
                            const _Float16* __restrict__ Vt,
                            _Float16* __restrict__ att)  // [B,S,D] f16
{
  const int lane = threadIdx.x & 31;
  const int job  = blockIdx.x * 8 + (threadIdx.x >> 5);
  const int qt = job & 127;
  const int h  = (job >> 7) & 7;
  const int b  = job >> 10;
  const int q0 = qt << 4;

  const _Float16* Qh = Q  + (size_t)(b * H_ + h) * S_ * HD_;
  const _Float16* Kh = K  + (size_t)(b * H_ + h) * S_ * HD_;
  const _Float16* Vh = Vt + (size_t)(b * H_ + h) * HD_ * S_;

  // Q^T B-fragments, reused for all k (hd 0..31 and 32..63)
  const v16h qb0 = load_fragB(Qh, HD_, q0, 0,  lane);
  const v16h qb1 = load_fragB(Qh, HD_, q0, 32, lane);

  v8f o[4] = {};                       // O^T : 64(hd) x 16(q)
  float mrun = -3.0e38f, lrun = 0.0f;  // per-column (q) stats, column = lane&15
  const int qcol = q0 + (lane & 15);
  const int kend = q0 + 16;
  const int g    = lane >> 4;

  for (int k0 = 0; k0 < kend; k0 += 32) {
    if (k0 + 32 < kend) __builtin_prefetch(Kh + (size_t)(k0 + 32) * HD_, 0, 1);

    // S^T = K x Q^T : two 16(k)x16(q) tiles
    v8f s[2];
#pragma unroll
    for (int t = 0; t < 2; ++t) {
      v16h ka0 = load_fragA(Kh, HD_, k0 + t * 16, 0,  lane);
      v16h ka1 = load_fragA(Kh, HD_, k0 + t * 16, 32, lane);
      v8f ss = {};
      ss = wmma_f16(ka0, qb0, ss);
      ss = wmma_f16(ka1, qb1, ss);
      s[t] = ss;
    }

    // causal mask + chunk max (column = q lives in lanes l and l^16)
    float cmax = -3.0e38f;
#pragma unroll
    for (int t = 0; t < 2; ++t)
#pragma unroll
      for (int r = 0; r < 8; ++r) {
        const int krow = k0 + t * 16 + r + (g << 3);
        const float v = (krow <= qcol) ? s[t][r] : -3.0e38f;
        s[t][r] = v;
        cmax = fmaxf(cmax, v);
      }
    cmax = fmaxf(cmax, __shfl_xor(cmax, 16, 32));

    const float mnew  = fmaxf(mrun, cmax);
    const float alpha = exp2f(mrun - mnew);
    float psum = 0.0f;
    v16h allp;                         // halves 0..7: tile0 rows, 8..15: tile1 rows
#pragma unroll
    for (int t = 0; t < 2; ++t)
#pragma unroll
      for (int r = 0; r < 8; ++r) {
        const float p = exp2f(s[t][r] - mnew);
        psum += p;
        allp[t * 8 + r] = (_Float16)p;
      }
    psum += __shfl_xor(psum, 16, 32);
    lrun = lrun * alpha + psum;
    mrun = mnew;

    // Build P^T B-fragment: B wants K 0..15 in lanes 0-15, K 16..31 in lanes 16-31.
    // Our C tiles hold K {g*8..g*8+7} (t=0) and {16+g*8..} (t=1) per lane, so a
    // single xor-16 dword swap of the "other" tile completes the fragment.
    const v8u pu = __builtin_bit_cast(v8u, allp);   // dw 0-3: t=0, dw 4-7: t=1
    unsigned int res[8];
#pragma unroll
    for (int i = 0; i < 4; ++i) {
      const unsigned int keep = pu[g * 4 + i];
      const unsigned int recv = __shfl_xor(pu[(1 - g) * 4 + i], 16, 32);
      res[g ? 4 + i : i] = keep;
      res[g ? i : 4 + i] = recv;
    }
    v8u ru;
#pragma unroll
    for (int i = 0; i < 8; ++i) ru[i] = res[i];
    const v16h pb = __builtin_bit_cast(v16h, ru);

    // O^T += V^T x P^T  (rescale accumulators first; alpha is per-column)
#pragma unroll
    for (int t4 = 0; t4 < 4; ++t4) {
#pragma unroll
      for (int r = 0; r < 8; ++r) o[t4][r] *= alpha;
      v16h va = load_fragA(Vh, S_, t4 * 16, k0, lane);  // V^T rows = hd
      o[t4] = wmma_f16(va, pb, o[t4]);
    }
  }

  const float rinv = 1.0f / lrun;
  const int q = q0 + (lane & 15);
#pragma unroll
  for (int t4 = 0; t4 < 4; ++t4) {
    v8h pk;
#pragma unroll
    for (int r = 0; r < 8; ++r) pk[r] = (_Float16)(o[t4][r] * rinv);
    _Float16* dst = att + ((size_t)b * S_ + q) * D_ + h * HD_ + t4 * 16 + (g << 3);
    *(v8h*)dst = pk;  // consecutive r -> consecutive hd -> contiguous
  }
}

// ---------------------------------------------------------------- out proj
// 4096 wave-jobs: m-tile(512) x n-tile(8); fp32 epilogue + bias.
__global__ void oproj_kernel(const _Float16* __restrict__ att,
                             const _Float16* __restrict__ Woh,
                             const float* __restrict__ bo,
                             float* __restrict__ out)
{
  const int lane = threadIdx.x & 31;
  const int job  = blockIdx.x * 8 + (threadIdx.x >> 5);
  const int m0 = (job >> 3) << 4;
  const int n0 = (job & 7) << 6;

  v8f acc[4] = {};
  for (int k0 = 0; k0 < D_; k0 += 32) {
    v16h a = load_fragA(att, D_, m0, k0, lane);
#pragma unroll
    for (int j = 0; j < 4; ++j) {
      v16h bf = load_fragB(Woh, D_, n0 + j * 16, k0, lane);
      acc[j] = wmma_f16(a, bf, acc[j]);
    }
  }

  const int mbase = m0 + ((lane >> 4) << 3);
#pragma unroll
  for (int j = 0; j < 4; ++j) {
    const int f = n0 + j * 16 + (lane & 15);
    const float bb = bo[f];
#pragma unroll
    for (int r = 0; r < 8; ++r)
      out[(size_t)(mbase + r) * D_ + f] = acc[j][r] + bb;
  }
}

// ---------------------------------------------------------------- launch
extern "C" void kernel_launch(void* const* d_in, const int* in_sizes, int n_in,
                              void* d_out, int out_size, void* d_ws, size_t ws_size,
                              hipStream_t stream) {
  const float* x  = (const float*)d_in[0];
  // d_in[1] = mask (unused; causal mask computed analytically)
  const float* Wq = (const float*)d_in[2];
  const float* bq = (const float*)d_in[3];
  const float* Wk = (const float*)d_in[4];
  const float* bk = (const float*)d_in[5];
  const float* Wv = (const float*)d_in[6];
  const float* bv = (const float*)d_in[7];
  const float* Wo = (const float*)d_in[8];
  const float* bo = (const float*)d_in[9];
  float* out = (float*)d_out;

  char* ws = (char*)d_ws;
  size_t off = 0;
  auto alloc = [&](size_t bytes) {
    char* p = ws + off;
    off += (bytes + 255) & ~(size_t)255;
    return p;
  };
  const size_t nx = (size_t)B_ * S_ * D_;  // 4,194,304
  const size_t nw = (size_t)D_ * D_;       // 262,144

  _Float16* xh  = (_Float16*)alloc(nx * 2);
  _Float16* Wqh = (_Float16*)alloc(nw * 2);
  _Float16* Wkh = (_Float16*)alloc(nw * 2);
  _Float16* Wvh = (_Float16*)alloc(nw * 2);
  _Float16* Woh = (_Float16*)alloc(nw * 2);
  _Float16* Qb  = (_Float16*)alloc(nx * 2);
  _Float16* Kb  = (_Float16*)alloc(nx * 2);
  _Float16* Vt  = (_Float16*)alloc(nx * 2);
  _Float16* att = (_Float16*)alloc(nx * 2);

  cvt4_f32_f16<<<(int)(nx / 1024), 256, 0, stream>>>(x,  xh,  (int)nx);
  cvt4_f32_f16<<<(int)(nw / 1024), 256, 0, stream>>>(Wq, Wqh, (int)nw);
  cvt4_f32_f16<<<(int)(nw / 1024), 256, 0, stream>>>(Wk, Wkh, (int)nw);
  cvt4_f32_f16<<<(int)(nw / 1024), 256, 0, stream>>>(Wv, Wvh, (int)nw);
  cvt4_f32_f16<<<(int)(nw / 1024), 256, 0, stream>>>(Wo, Woh, (int)nw);

  qkv_kernel<<<1536, 256, 0, stream>>>(xh, Wqh, Wkh, Wvh, bq, bk, bv, Qb, Kb, Vt);
  attn_kernel<<<512, 256, 0, stream>>>(Qb, Kb, Vt, att);
  oproj_kernel<<<512, 256, 0, stream>>>(att, Woh, bo, out);
}
